// AtomfeatsToTorsion_7361573945695
// MI455X (gfx1250) — compile-verified, matrix operand order checked by reading, not run on hardware
//
#include <hip/hip_runtime.h>
#include <math.h>

typedef __attribute__((ext_vector_type(16))) _Float16 v16h;
typedef __attribute__((ext_vector_type(8)))  _Float16 v8h;
typedef __attribute__((ext_vector_type(8)))  float    v8f;
typedef __attribute__((ext_vector_type(4)))  float    v4f;

#define DIM   128
#define NHEAD 8
#define HDIM  16
#define DDIM  32
#define NBND  10000
#define NEDG  320000

// ---------------------------------------------------------------------------
// Weight packing: f32 row-major W[K][N] -> f16 fragments in the exact
// v_wmma_f32_16x16x32_f16 B-operand lane layout, so each lane loads one
// contiguous 32-byte v16h.  Layout: P[((nt*numKt + kt)*32 + lane)*16 + e]
// ---------------------------------------------------------------------------
__global__ void aftt_pack_w(const float* __restrict__ W, _Float16* __restrict__ P,
                            int K, int N) {
  int idx = blockIdx.x * blockDim.x + threadIdx.x;
  if (idx >= K * N) return;
  int e    = idx & 15;
  int l    = (idx >> 4) & 31;
  int rest = idx >> 9;
  int numKt = K >> 5;
  int kt = rest % numKt;
  int nt = rest / numKt;
  int n  = nt * 16 + (l & 15);
  int off = (l < 16) ? 0 : 8;
  int k = kt * 32 + off + ((e < 8) ? e : (e - 8 + 16));
  P[idx] = (_Float16)W[(long)k * N + n];
}

// A fragment (16-bit 16x32 layout): lane&15 = row M, lane>=16 takes K+8 halves.
__device__ __forceinline__ v16h aftt_load_a(const _Float16* A, long rowBase, int kb, int lane) {
  int off = (lane & 16) ? 8 : 0;
  const v8h* p0 = (const v8h*)(A + rowBase + kb + off);
  const v8h* p1 = (const v8h*)(A + rowBase + kb + off + 16);
  v8h a0 = *p0, a1 = *p1;
  v16h r;
#pragma unroll
  for (int i = 0; i < 8; ++i) { r[i] = a0[i]; r[i + 8] = a1[i]; }
  return r;
}

__device__ __forceinline__ v16h aftt_load_b(const _Float16* P, int numKt, int nt, int kt, int lane) {
  return *(const v16h*)(P + ((((long)nt * numKt + kt) * 32 + lane) << 4));
}

// ---------------------------------------------------------------------------
// A-stationary tiled GEMM: one wave owns one 16-row M tile; the NUMKT A
// fragments live in registers and are reused across all 8 N tiles (only B
// fragments stream from the packed weights).  f32 accumulation via
// v_wmma_f32_16x16x32_f16.  Optional exact-GELU epilogue + f16 shadow copy.
// ---------------------------------------------------------------------------
template <int NUMKT>
__global__ void aftt_gemm(const _Float16* __restrict__ A, const _Float16* __restrict__ BP,
                          const float* __restrict__ bias, float biasScale,
                          float* __restrict__ C, _Float16* __restrict__ C16,
                          int doGelu) {
  const int lane = threadIdx.x;
  const int mt = blockIdx.x;
  const int K = NUMKT * 32;
  const long rowBase = (long)(mt * 16 + (lane & 15)) * K;
  v16h afr[NUMKT];
#pragma unroll
  for (int kt = 0; kt < NUMKT; ++kt) afr[kt] = aftt_load_a(A, rowBase, kt * 32, lane);

  const int mAdd = (lane & 16) ? 8 : 0;
#pragma unroll
  for (int nt = 0; nt < 8; ++nt) {
    v8f acc = {};
#pragma unroll
    for (int kt = 0; kt < NUMKT; ++kt) {
      v16h b = aftt_load_b(BP, NUMKT, nt, kt, lane);
      acc = __builtin_amdgcn_wmma_f32_16x16x32_f16(false, afr[kt], false, b, (short)0, acc, false, false);
    }
    const int col = nt * 16 + (lane & 15);
    const float bb = bias[col] * biasScale;
#pragma unroll
    for (int r = 0; r < 8; ++r) {
      long row = mt * 16 + mAdd + r;
      float x = acc[r] + bb;
      if (doGelu) x = 0.5f * x * (1.0f + erff(x * 0.70710678118654752f));
      C[row * DIM + col] = x;
      if (C16) C16[row * DIM + col] = (_Float16)x;
    }
  }
}

// ---------------------------------------------------------------------------
// Small prep kernels
// ---------------------------------------------------------------------------
__global__ void aftt_rbpos(const float* __restrict__ coords, const int* __restrict__ rbi,
                           float* __restrict__ rb_pos) {
  int idx = blockIdx.x * blockDim.x + threadIdx.x;
  if (idx >= NBND * 3) return;
  int b = idx / 3, c = idx % 3;
  int a0 = rbi[b], a1 = rbi[NBND + b];
  rb_pos[idx] = 0.5f * (coords[(long)a0 * 3 + c] + coords[(long)a1 * 3 + c]);
}

// dsum = dih_fwd + dih_bwd = [fi+fl, fj+fk, fj+fk, fi+fl]  (f16, [NB,512])
__global__ void aftt_dsum(const float* __restrict__ atom_feats, const int* __restrict__ tt,
                          _Float16* __restrict__ dsum) {
  int idx = blockIdx.x * blockDim.x + threadIdx.x;
  if (idx >= NBND * DIM) return;
  int b = idx >> 7, c = idx & 127;
  int i = tt[b * 4 + 0], j = tt[b * 4 + 1], k = tt[b * 4 + 2], l = tt[b * 4 + 3];
  _Float16 h0 = (_Float16)(atom_feats[(long)i * DIM + c] + atom_feats[(long)l * DIM + c]);
  _Float16 h1 = (_Float16)(atom_feats[(long)j * DIM + c] + atom_feats[(long)k * DIM + c]);
  long rb = (long)b * 512;
  dsum[rb + c] = h0; dsum[rb + 128 + c] = h1; dsum[rb + 256 + c] = h1; dsum[rb + 384 + c] = h0;
}

__global__ void aftt_f32_to_f16(const float* __restrict__ s, _Float16* __restrict__ d, int n) {
  int i = blockIdx.x * blockDim.x + threadIdx.x;
  if (i < n) d[i] = (_Float16)s[i];
}

// ---------------------------------------------------------------------------
// Edge tile builder: 16 edges -> LDS atom_input [16][160] f16 (128 feats + 32
// Gaussian smearing values recomputed from coords/rb_pos).
// ---------------------------------------------------------------------------
__device__ __forceinline__ void aftt_build_tile(int tile,
    const float* __restrict__ atom_feats, const float* __restrict__ coords,
    const float* __restrict__ rb_pos, const int* __restrict__ edge_src,
    const int* __restrict__ edge_tgt, _Float16 (*Ain)[160], int* tgts, int lane) {
  int eL = lane & 15;
  int e = tile * 16 + eL;
  int src = edge_src[e];
  int tgt = edge_tgt[e];
  if (lane < 16) tgts[eL] = tgt;
  const float* af = atom_feats + (long)src * DIM;
  if (lane < 16) {
#pragma unroll
    for (int c = 0; c < 80; c += 4) {
      v4f f = *(const v4f*)(af + c);
      Ain[eL][c + 0] = (_Float16)f[0]; Ain[eL][c + 1] = (_Float16)f[1];
      Ain[eL][c + 2] = (_Float16)f[2]; Ain[eL][c + 3] = (_Float16)f[3];
    }
  } else {
#pragma unroll
    for (int c = 80; c < 128; c += 4) {
      v4f f = *(const v4f*)(af + c);
      Ain[eL][c + 0] = (_Float16)f[0]; Ain[eL][c + 1] = (_Float16)f[1];
      Ain[eL][c + 2] = (_Float16)f[2]; Ain[eL][c + 3] = (_Float16)f[3];
    }
    float dx = coords[(long)src * 3 + 0] - rb_pos[(long)tgt * 3 + 0];
    float dy = coords[(long)src * 3 + 1] - rb_pos[(long)tgt * 3 + 1];
    float dz = coords[(long)src * 3 + 2] - rb_pos[(long)tgt * 3 + 2];
    float d = sqrtf(dx * dx + dy * dy + dz * dz);
    const float delta = 5.0f / 31.0f;
    const float coeff = -0.5f / (delta * delta);
#pragma unroll
    for (int g = 0; g < DDIM; ++g) {
      float t = d - delta * (float)g;
      Ain[eL][128 + g] = (_Float16)expf(coeff * t * t);
    }
  }
}

// monotonic float<->uint mapping for atomicMax-based segment max
__device__ __forceinline__ unsigned aftt_fmap(float f) {
  unsigned b = __float_as_uint(f);
  return b ^ ((b & 0x80000000u) ? 0xFFFFFFFFu : 0x80000000u);
}
__device__ __forceinline__ float aftt_funmap(unsigned k) {
  unsigned b = (k & 0x80000000u) ? (k ^ 0x80000000u) : ~k;
  return __uint_as_float(b);
}

// ---------------------------------------------------------------------------
// Edge pass 1: K projection via WMMA (A-stationary: 5 fragments in registers,
// reused over all 8 head tiles), per-(edge,head) logits, segment max.
// ---------------------------------------------------------------------------
__global__ void aftt_edge_logits(const float* __restrict__ atom_feats,
    const float* __restrict__ coords, const float* __restrict__ rb_pos,
    const int* __restrict__ edge_src, const int* __restrict__ edge_tgt,
    const _Float16* __restrict__ WkP, const float* __restrict__ bk,
    const float* __restrict__ q, float* __restrict__ logits,
    unsigned* __restrict__ segmax) {
  __shared__ __align__(16) _Float16 Ain[16][160];
  __shared__ float Kt[16][128];
  __shared__ int tgts[16];
  const int lane = threadIdx.x;
  const int tile = blockIdx.x;
  aftt_build_tile(tile, atom_feats, coords, rb_pos, edge_src, edge_tgt, Ain, tgts, lane);
  __syncthreads();
  const long rowBase = (long)(lane & 15) * 160;
  v16h afr[5];
#pragma unroll
  for (int kt = 0; kt < 5; ++kt) afr[kt] = aftt_load_a(&Ain[0][0], rowBase, kt * 32, lane);
  const int mAdd = (lane & 16) ? 8 : 0;
#pragma unroll
  for (int nt = 0; nt < 8; ++nt) {
    v8f acc = {};
#pragma unroll
    for (int kt = 0; kt < 5; ++kt) {
      v16h b = aftt_load_b(WkP, 5, nt, kt, lane);
      acc = __builtin_amdgcn_wmma_f32_16x16x32_f16(false, afr[kt], false, b, (short)0, acc, false, false);
    }
    int col = nt * 16 + (lane & 15);
    float bb = bk[col];
#pragma unroll
    for (int r = 0; r < 8; ++r) Kt[mAdd + r][col] = acc[r] + bb;
  }
  __syncthreads();
  for (int p = lane; p < 128; p += 32) {
    int eL = p >> 3, h = p & 7;
    int t = tgts[eL];
    const float* qp = q + (long)t * DIM + h * HDIM;
    float s = 0.f;
#pragma unroll
    for (int c = 0; c < HDIM; ++c) s += qp[c] * Kt[eL][h * HDIM + c];
    s *= 0.25f;  // 1/sqrt(HEAD_DIM)
    long e = (long)tile * 16 + eL;
    logits[e * NHEAD + h] = s;
    atomicMax(&segmax[t * NHEAD + h], aftt_fmap(s));
  }
}

// exp(l - segmax) + segment sum
__global__ void aftt_expsum(const int* __restrict__ edge_tgt,
    const unsigned* __restrict__ segmax, float* __restrict__ logits,
    float* __restrict__ segsum) {
  long i = (long)blockIdx.x * blockDim.x + threadIdx.x;
  if (i >= (long)NEDG * NHEAD) return;
  long e = i >> 3; int h = (int)(i & 7);
  int t = edge_tgt[e];
  float m = aftt_funmap(segmax[t * NHEAD + h]);
  float ex = expf(logits[i] - m);
  logits[i] = ex;
  atomicAdd(&segsum[t * NHEAD + h], ex);
}

// ---------------------------------------------------------------------------
// Edge pass 2: V projection via WMMA (A-stationary), attention-weighted
// scatter-add into out[NB,128].
// ---------------------------------------------------------------------------
__global__ void aftt_edge_out(const float* __restrict__ atom_feats,
    const float* __restrict__ coords, const float* __restrict__ rb_pos,
    const int* __restrict__ edge_src, const int* __restrict__ edge_tgt,
    const _Float16* __restrict__ WvP, const float* __restrict__ bv,
    const float* __restrict__ logits, const float* __restrict__ segsum,
    float* __restrict__ outacc) {
  __shared__ __align__(16) _Float16 Ain[16][160];
  __shared__ float attn[16][NHEAD];
  __shared__ int tgts[16];
  const int lane = threadIdx.x;
  const int tile = blockIdx.x;
  aftt_build_tile(tile, atom_feats, coords, rb_pos, edge_src, edge_tgt, Ain, tgts, lane);
  __syncthreads();
  for (int p = lane; p < 128; p += 32) {
    int eL = p >> 3, h = p & 7;
    long e = (long)tile * 16 + eL;
    int t = tgts[eL];
    attn[eL][h] = logits[e * NHEAD + h] / (segsum[t * NHEAD + h] + 1e-16f);
  }
  __syncthreads();
  const long rowBase = (long)(lane & 15) * 160;
  v16h afr[5];
#pragma unroll
  for (int kt = 0; kt < 5; ++kt) afr[kt] = aftt_load_a(&Ain[0][0], rowBase, kt * 32, lane);
  const int mAdd = (lane & 16) ? 8 : 0;
#pragma unroll
  for (int nt = 0; nt < 8; ++nt) {   // head == nt (HEAD_DIM==16)
    v8f acc = {};
#pragma unroll
    for (int kt = 0; kt < 5; ++kt) {
      v16h b = aftt_load_b(WvP, 5, nt, kt, lane);
      acc = __builtin_amdgcn_wmma_f32_16x16x32_f16(false, afr[kt], false, b, (short)0, acc, false, false);
    }
    int col = nt * 16 + (lane & 15);
    float bb = bv[col];
#pragma unroll
    for (int r = 0; r < 8; ++r) {
      int m = mAdd + r;
      float val = (acc[r] + bb) * attn[m][nt];
      atomicAdd(&outacc[(long)tgts[m] * DIM + col], val);
    }
  }
}

// final 128 -> 2 projection
__global__ void aftt_final(const float* __restrict__ hbuf, const float* __restrict__ Wt2,
                           const float* __restrict__ bt2, float* __restrict__ out) {
  int b = blockIdx.x * blockDim.x + threadIdx.x;
  if (b >= NBND) return;
  float s0 = bt2[0], s1 = bt2[1];
  const float* hr = hbuf + (long)b * DIM;
#pragma unroll 4
  for (int c = 0; c < DIM; ++c) {
    float x = hr[c];
    s0 += x * Wt2[c * 2 + 0];
    s1 += x * Wt2[c * 2 + 1];
  }
  out[b * 2 + 0] = s0;
  out[b * 2 + 1] = s1;
}

// ---------------------------------------------------------------------------
extern "C" void kernel_launch(void* const* d_in, const int* in_sizes, int n_in,
                              void* d_out, int out_size, void* d_ws, size_t ws_size,
                              hipStream_t stream) {
  const float* atom_feats = (const float*)d_in[0];
  const float* coords     = (const float*)d_in[1];
  const int*   rbi        = (const int*)d_in[2];
  const int*   edge_tgt   = (const int*)d_in[3];
  const int*   edge_src   = (const int*)d_in[4];
  const int*   tt         = (const int*)d_in[5];
  const float* W_dih = (const float*)d_in[6];
  const float* b_dih = (const float*)d_in[7];
  const float* Wq    = (const float*)d_in[8];
  const float* bq    = (const float*)d_in[9];
  const float* Wk    = (const float*)d_in[10];
  const float* bk    = (const float*)d_in[11];
  const float* Wv    = (const float*)d_in[12];
  const float* bv    = (const float*)d_in[13];
  const float* Wout  = (const float*)d_in[14];
  const float* bout  = (const float*)d_in[15];
  const float* Wt1   = (const float*)d_in[16];
  const float* bt1   = (const float*)d_in[17];
  const float* Wt2   = (const float*)d_in[18];
  const float* bt2   = (const float*)d_in[19];
  float* out = (float*)d_out;

  size_t cur = 0;
  auto walloc = [&](size_t bytes) -> void* {
    void* p = (char*)d_ws + cur;
    cur += (bytes + 255) & ~(size_t)255;
    return p;
  };
  _Float16* WdihP = (_Float16*)walloc((size_t)512 * 128 * 2);
  _Float16* WqP   = (_Float16*)walloc((size_t)128 * 128 * 2);
  _Float16* WkP   = (_Float16*)walloc((size_t)160 * 128 * 2);
  _Float16* WvP   = (_Float16*)walloc((size_t)160 * 128 * 2);
  _Float16* WoutP = (_Float16*)walloc((size_t)128 * 128 * 2);
  _Float16* Wt1P  = (_Float16*)walloc((size_t)128 * 128 * 2);
  float*    rb_pos   = (float*)walloc((size_t)NBND * 3 * 4);
  _Float16* dsum     = (_Float16*)walloc((size_t)NBND * 512 * 2);
  float*    rbf_f32  = (float*)walloc((size_t)NBND * DIM * 4);
  _Float16* rbf_h    = (_Float16*)walloc((size_t)NBND * DIM * 2);
  float*    q_f32    = (float*)walloc((size_t)NBND * DIM * 4);
  float*    logits   = (float*)walloc((size_t)NEDG * NHEAD * 4);
  unsigned* segmax   = (unsigned*)walloc((size_t)NBND * NHEAD * 4);
  float*    segsum   = (float*)walloc((size_t)NBND * NHEAD * 4);
  float*    outacc   = (float*)walloc((size_t)NBND * DIM * 4);
  _Float16* out_h    = (_Float16*)walloc((size_t)NBND * DIM * 2);
  float*    out2_f32 = (float*)walloc((size_t)NBND * DIM * 4);
  _Float16* out2_h   = (_Float16*)walloc((size_t)NBND * DIM * 2);
  float*    h_f32    = (float*)walloc((size_t)NBND * DIM * 4);
  (void)ws_size; (void)in_sizes; (void)n_in; (void)out_size;

  // --- weight packing ---
  auto packGrid = [](int k, int n) { return dim3((k * n + 255) / 256); };
  aftt_pack_w<<<packGrid(512, 128), 256, 0, stream>>>(W_dih, WdihP, 512, 128);
  aftt_pack_w<<<packGrid(128, 128), 256, 0, stream>>>(Wq,    WqP,   128, 128);
  aftt_pack_w<<<packGrid(160, 128), 256, 0, stream>>>(Wk,    WkP,   160, 128);
  aftt_pack_w<<<packGrid(160, 128), 256, 0, stream>>>(Wv,    WvP,   160, 128);
  aftt_pack_w<<<packGrid(128, 128), 256, 0, stream>>>(Wout,  WoutP, 128, 128);
  aftt_pack_w<<<packGrid(128, 128), 256, 0, stream>>>(Wt1,   Wt1P,  128, 128);

  // --- prep ---
  aftt_rbpos<<<(NBND * 3 + 255) / 256, 256, 0, stream>>>(coords, rbi, rb_pos);
  aftt_dsum<<<(NBND * DIM + 255) / 256, 256, 0, stream>>>(atom_feats, tt, dsum);
  hipMemsetAsync(segmax, 0, (size_t)NBND * NHEAD * 4, stream);
  hipMemsetAsync(segsum, 0, (size_t)NBND * NHEAD * 4, stream);
  hipMemsetAsync(outacc, 0, (size_t)NBND * DIM * 4, stream);

  // --- rb_feats = dsum @ W_dih + 2*b_dih ; q = rb_feats @ Wq + bq ---
  dim3 blk(32);
  aftt_gemm<16><<<dim3(NBND / 16), blk, 0, stream>>>(dsum, WdihP, b_dih, 2.0f,
                                                     rbf_f32, rbf_h, 0);
  aftt_gemm<4><<<dim3(NBND / 16), blk, 0, stream>>>(rbf_h, WqP, bq, 1.0f,
                                                    q_f32, nullptr, 0);

  // --- edge passes (fused K/V projection + attention) ---
  aftt_edge_logits<<<NEDG / 16, blk, 0, stream>>>(atom_feats, coords, rb_pos,
      edge_src, edge_tgt, WkP, bk, q_f32, logits, segmax);
  aftt_expsum<<<(int)(((long)NEDG * NHEAD + 255) / 256), 256, 0, stream>>>(
      edge_tgt, segmax, logits, segsum);
  aftt_edge_out<<<NEDG / 16, blk, 0, stream>>>(atom_feats, coords, rb_pos,
      edge_src, edge_tgt, WvP, bv, logits, segsum, outacc);

  // --- output MLP ---
  aftt_f32_to_f16<<<(NBND * DIM + 255) / 256, 256, 0, stream>>>(outacc, out_h, NBND * DIM);
  aftt_gemm<4><<<dim3(NBND / 16), blk, 0, stream>>>(out_h, WoutP, bout, 1.0f,
                                                    out2_f32, out2_h, 0);
  aftt_gemm<4><<<dim3(NBND / 16), blk, 0, stream>>>(out2_h, Wt1P, bt1, 1.0f,
                                                    h_f32, nullptr, 1);
  aftt_final<<<(NBND + 255) / 256, 256, 0, stream>>>(h_f32, Wt2, bt2, out);
}
